// MHA_block_22368189678066
// MI455X (gfx1250) — compile-verified, matrix operand order checked by reading, not run on hardware
//
#include <hip/hip_runtime.h>
#include <hip/hip_bf16.h>

// ---------------------------------------------------------------------------
// Types for CDNA5 WMMA (wave32): V_WMMA_F32_16X16X32_BF16
// ---------------------------------------------------------------------------
typedef __bf16 bf16_t;
typedef __attribute__((ext_vector_type(16))) __bf16 v16bf;
typedef __attribute__((ext_vector_type(8)))  float  v8f;

#define WMMA_BF16(a, b, c) \
    __builtin_amdgcn_wmma_f32_16x16x32_bf16(false, (a), false, (b), (short)0, (c), false, false)

// Model constants
static constexpr int Bc = 4, Sc = 8192, Dc = 768, Hc = 12, DHc = 64, FFc = 3072;
static constexpr int BS = Bc * Sc;

// ---------------------------------------------------------------------------
// Async global->LDS copy (CDNA5): per-lane 16B, tracked by ASYNCcnt.
// ---------------------------------------------------------------------------
__device__ __forceinline__ void async_copy16(bf16_t* lds, const bf16_t* g) {
    unsigned        l  = (unsigned)(uintptr_t)lds;          // LDS byte offset (addr[31:0])
    unsigned long long ga = (unsigned long long)(uintptr_t)g;
    asm volatile("global_load_async_to_lds_b128 %0, %1, off"
                 :: "v"(l), "v"(ga) : "memory");
}
__device__ __forceinline__ void async_wait0() {
    asm volatile("s_wait_asynccnt 0" ::: "memory");
}

// ---------------------------------------------------------------------------
// Fragment helpers.
// ISA 16-bit A-matrix 16x32 layout (wave32):
//   lanes 0-15  : row M = lane,    elems = K{0..7} then K{16..23}
//   lanes 16-31 : row M = lane-16, elems = K{8..15} then K{24..31}
// B uses the identical per-lane K ordering with lane selecting column N,
// so both A and B fragments are two contiguous 16-byte reads when the
// source tile is K-contiguous.
// ---------------------------------------------------------------------------
__device__ __forceinline__ v16bf frag_load_contig(const bf16_t* p) {
    union { uint4 u; __bf16 h[8]; } lo, hi;
    lo.u = *(const uint4*)(p);
    hi.u = *(const uint4*)(p + 16);
    v16bf f;
#pragma unroll
    for (int i = 0; i < 8; ++i) { f[i] = lo.h[i]; f[i + 8] = hi.h[i]; }
    return f;
}

__device__ __forceinline__ const bf16_t* frag_ptr(const bf16_t* base, int ld,
                                                  int row0, int k0, int lane) {
    int r    = lane & 15;
    int half = (lane >> 4) * 8;
    return base + (size_t)(row0 + r) * ld + k0 + half;
}

// ---------------------------------------------------------------------------
// Weight convert + transpose: W[K][N] f32  ->  WT[N][K] bf16 (32x32 LDS tiles)
// ---------------------------------------------------------------------------
__global__ __launch_bounds__(256) void conv_transpose_kernel(
    const float* __restrict__ in, bf16_t* __restrict__ out, int K, int N) {
    __shared__ float tile[32][33];
    const int k0 = blockIdx.y * 32, n0 = blockIdx.x * 32;
    const int tx = threadIdx.x & 31, ty = threadIdx.x >> 5;  // 32 x 8
#pragma unroll
    for (int i = ty; i < 32; i += 8)
        tile[i][tx] = in[(size_t)(k0 + i) * N + n0 + tx];
    __syncthreads();
#pragma unroll
    for (int i = ty; i < 32; i += 8)
        out[(size_t)(n0 + i) * K + k0 + tx] = (bf16_t)tile[tx][i];
}

// ---------------------------------------------------------------------------
// LayerNorm: one 256-thread block per row of D=768, bf16 output
// ---------------------------------------------------------------------------
__global__ __launch_bounds__(256) void ln_kernel(const float* __restrict__ x,
                                                 const float* __restrict__ g,
                                                 const float* __restrict__ be,
                                                 bf16_t* __restrict__ out, int D) {
    __shared__ float red[256];
    const int row  = blockIdx.x;
    const float* xr = x + (size_t)row * D;

    float s = 0.f;
    for (int i = threadIdx.x; i < D; i += 256) s += xr[i];
    red[threadIdx.x] = s; __syncthreads();
    for (int st = 128; st > 0; st >>= 1) {
        if (threadIdx.x < st) red[threadIdx.x] += red[threadIdx.x + st];
        __syncthreads();
    }
    const float mean = red[0] / (float)D;
    __syncthreads();

    float vs = 0.f;
    for (int i = threadIdx.x; i < D; i += 256) { float d = xr[i] - mean; vs += d * d; }
    red[threadIdx.x] = vs; __syncthreads();
    for (int st = 128; st > 0; st >>= 1) {
        if (threadIdx.x < st) red[threadIdx.x] += red[threadIdx.x + st];
        __syncthreads();
    }
    const float rstd = rsqrtf(red[0] / (float)D + 1e-5f);
    __syncthreads();

    for (int i = threadIdx.x; i < D; i += 256)
        out[(size_t)row * D + i] = (bf16_t)((xr[i] - mean) * rstd * g[i] + be[i]);
}

// ---------------------------------------------------------------------------
// Generic tiled WMMA GEMM: C[M,N] = act(A[M,K] @ WT[N][K]^T + bias) (+ resid)
// Block tile 128x128, 8 waves of 32x64 (2x4 WMMA tiles), K step 32.
// Double-buffered async global->LDS staging: iteration t computes from
// buffer t&1 while ASYNCcnt-tracked copies fill buffer (t+1)&1; the wait +
// single barrier sits after the WMMA block, so copy latency hides under
// matrix math.
// ---------------------------------------------------------------------------
__global__ __launch_bounds__(256) void gemm_bias_act_kernel(
    const bf16_t* __restrict__ A, const bf16_t* __restrict__ WT,
    const float* __restrict__ bias, const float* __restrict__ resid,
    float* __restrict__ outF, bf16_t* __restrict__ outH,
    int M, int N, int K, int act) {
    constexpr int BM = 128, BN = 128, BK = 32;
    __shared__ bf16_t As[2 * BM * BK];  // double-buffered [128][32]
    __shared__ bf16_t Bs[2 * BN * BK];  // double-buffered [128][32] (rows = N cols)

    const int tid  = threadIdx.x;
    const int lane = tid & 31;
    const int wave = tid >> 5;
    const int m0 = blockIdx.y * BM;
    const int n0 = blockIdx.x * BN;
    const int waveM = (wave & 3) * 32;   // 0..96
    const int waveN = (wave >> 2) * 64;  // 0 / 64

    const int row = tid >> 1;          // 0..127
    const int kc  = (tid & 1) * 16;    // 0 / 16

    auto stage = [&](int k0, int buf) {
        const bf16_t* gA = &A [(size_t)(m0 + row) * K + k0 + kc];
        const bf16_t* gB = &WT[(size_t)(n0 + row) * K + k0 + kc];
        bf16_t* as = As + buf * (BM * BK);
        bf16_t* bs = Bs + buf * (BN * BK);
        async_copy16(&as[row * BK + kc],     gA);
        async_copy16(&as[row * BK + kc + 8], gA + 8);
        async_copy16(&bs[row * BK + kc],     gB);
        async_copy16(&bs[row * BK + kc + 8], gB + 8);
        __builtin_prefetch(gA + BK, 0, 1);
        __builtin_prefetch(gB + BK, 0, 1);
    };

    v8f acc[2][4] = {};
    const int T = K / BK;

    stage(0, 0);
    async_wait0();
    __syncthreads();

    for (int t = 0; t < T; ++t) {
        const bf16_t* as = As + (t & 1) * (BM * BK);
        const bf16_t* bs = Bs + (t & 1) * (BN * BK);
        if (t + 1 < T) stage((t + 1) * BK, (t + 1) & 1);

        v16bf af[2], bfrag[4];
#pragma unroll
        for (int i = 0; i < 2; ++i)
            af[i] = frag_load_contig(frag_ptr(as, BK, waveM + i * 16, 0, lane));
#pragma unroll
        for (int j = 0; j < 4; ++j)
            bfrag[j] = frag_load_contig(frag_ptr(bs, BK, waveN + j * 16, 0, lane));
#pragma unroll
        for (int i = 0; i < 2; ++i)
#pragma unroll
            for (int j = 0; j < 4; ++j)
                acc[i][j] = WMMA_BF16(af[i], bfrag[j], acc[i][j]);

        if (t + 1 < T) {
            async_wait0();
            __syncthreads();
        }
    }

    // Epilogue. C/D layout: lane n = lane&15; rows = (lane>=16 ? 8 : 0)+r
    const int nOut  = lane & 15;
    const int mBase = (lane >> 4) * 8;
#pragma unroll
    for (int i = 0; i < 2; ++i)
#pragma unroll
        for (int j = 0; j < 4; ++j) {
            const int gn = n0 + waveN + j * 16 + nOut;
            const float bval = bias ? bias[gn] : 0.f;
#pragma unroll
            for (int r = 0; r < 8; ++r) {
                const int gm = m0 + waveM + i * 16 + mBase + r;
                float v = acc[i][j][r] + bval;
                if (act) v = 0.5f * v * (1.0f + erff(v * 0.70710678118654752f));
                if (resid) v += resid[(size_t)gm * N + gn];
                if (outF) outF[(size_t)gm * N + gn] = v;
                else      outH[(size_t)gm * N + gn] = (bf16_t)v;
            }
        }
}

// ---------------------------------------------------------------------------
// relu/mask + cos-sin reweighting: qraw/kraw [BS,768] -> q2/k2 [B,S,H,128]
// ---------------------------------------------------------------------------
__global__ void reweight_kernel(const bf16_t* __restrict__ qraw,
                                const bf16_t* __restrict__ kraw,
                                const float* __restrict__ cosb,
                                const float* __restrict__ sinb,
                                const unsigned char* __restrict__ mask,
                                bf16_t* __restrict__ q2, bf16_t* __restrict__ k2,
                                int total) {
    int t = blockIdx.x * 256 + threadIdx.x;
    if (t >= total) return;                 // t = (bs*H + h)*64 + dh
    const int dh = t & 63;
    const int rest = t >> 6;
    const int h  = rest % Hc;
    const int bs = rest / Hc;

    const size_t rawIdx = (size_t)bs * Dc + h * DHc + dh;
    float q = (float)qraw[rawIdx]; q = q > 0.f ? q : 0.f;
    float k = (float)kraw[rawIdx]; k = k > 0.f ? k : 0.f;
    if (mask[bs]) k = 0.f;

    const float c = cosb[t], s = sinb[t];
    const size_t o = ((size_t)bs * Hc + h) * 128 + dh;
    q2[o]      = (bf16_t)(q * c);
    q2[o + 64] = (bf16_t)(q * s);
    k2[o]      = (bf16_t)(k * c);
    k2[o + 64] = (bf16_t)(k * s);
}

// ---------------------------------------------------------------------------
// kv[b,h,128,64] = sum_s k2[b,s,h,:]^T (x) v[b,s,h,:]   (+ ksum folded in)
// One block per (b,h). Tiles of 32 s staged transposed into LDS so all WMMA
// fragments are contiguous b128 LDS reads. Wave w owns kv rows [16w,16w+16).
// ---------------------------------------------------------------------------
__global__ __launch_bounds__(256) void kv_kernel(const bf16_t* __restrict__ k2,
                                                 const bf16_t* __restrict__ v,
                                                 bf16_t* __restrict__ kv,
                                                 float* __restrict__ ksum) {
    __shared__ bf16_t kT[128 * 32];  // [d][s]  8 KB
    __shared__ bf16_t vT[64 * 32];   // [m][s]  4 KB
    const int bh = blockIdx.x;
    const int b = bh / Hc, h = bh % Hc;
    const int tid = threadIdx.x, lane = tid & 31, wave = tid >> 5;

    const int sl = tid >> 3;         // 0..31 (s within tile)
    const int g8 = tid & 7;

    v8f acc[4] = {};
    float ks = 0.f;  // ksum partial for d = tid (threads 0..127)

    for (int s0 = 0; s0 < Sc; s0 += 32) {
        __syncthreads();
        {   // stage k2 tile transposed: 32 x 128 -> kT[d][s]
            const bf16_t* kp = k2 + ((size_t)(b * Sc + s0 + sl) * Hc + h) * 128 + g8 * 16;
            union { uint4 u; __bf16 hh[8]; } u0, u1;
            u0.u = *(const uint4*)kp;
            u1.u = *(const uint4*)(kp + 8);
#pragma unroll
            for (int j = 0; j < 8; ++j) {
                kT[(g8 * 16 + j) * 32 + sl]     = u0.hh[j];
                kT[(g8 * 16 + 8 + j) * 32 + sl] = u1.hh[j];
            }
        }
        {   // stage v tile transposed: 32 x 64 -> vT[m][s]
            const bf16_t* vp = v + (size_t)(b * Sc + s0 + sl) * Dc + h * DHc + g8 * 8;
            union { uint4 u; __bf16 hh[8]; } u2;
            u2.u = *(const uint4*)vp;
#pragma unroll
            for (int j = 0; j < 8; ++j)
                vT[(g8 * 8 + j) * 32 + sl] = u2.hh[j];
        }
        __syncthreads();

        v16bf a = frag_load_contig(frag_ptr(kT, 32, wave * 16, 0, lane));
        v16bf bfrag[4];
#pragma unroll
        for (int jt = 0; jt < 4; ++jt)
            bfrag[jt] = frag_load_contig(frag_ptr(vT, 32, jt * 16, 0, lane));
#pragma unroll
        for (int jt = 0; jt < 4; ++jt)
            acc[jt] = WMMA_BF16(a, bfrag[jt], acc[jt]);

        if (tid < 128) {  // ksum from staged tile (avoids a second k2 pass)
#pragma unroll 8
            for (int ss = 0; ss < 32; ++ss) ks += (float)kT[tid * 32 + ss];
        }
    }

    const int nOut  = lane & 15;
    const int mBase = (lane >> 4) * 8;
#pragma unroll
    for (int jt = 0; jt < 4; ++jt)
#pragma unroll
        for (int r = 0; r < 8; ++r) {
            const int d = wave * 16 + mBase + r;
            const int m = jt * 16 + nOut;
            kv[(size_t)bh * 8192 + d * 64 + m] = (bf16_t)acc[jt][r];
        }
    if (tid < 128) ksum[(size_t)bh * 128 + tid] = ks;
}

// ---------------------------------------------------------------------------
// z[b,s,h] = 1 / (q2[b,s,h,:] . ksum[b,h,:] + eps)
// ---------------------------------------------------------------------------
__global__ void z_kernel(const bf16_t* __restrict__ q2,
                         const float* __restrict__ ksum,
                         float* __restrict__ z, int total) {
    int t = blockIdx.x * 256 + threadIdx.x;
    if (t >= total) return;                 // t = (b*S+s)*H + h
    const int h  = t % Hc;
    const int bs = t / Hc;
    const uint4* qp4 = (const uint4*)(q2 + (size_t)t * 128);
    const float* kp  = ksum + ((size_t)(bs / Sc) * Hc + h) * 128;
    float acc = 0.f;
#pragma unroll
    for (int c = 0; c < 16; ++c) {
        union { uint4 u; __bf16 hh[8]; } uu;
        uu.u = qp4[c];
#pragma unroll
        for (int j = 0; j < 8; ++j) acc += (float)uu.hh[j] * kp[c * 8 + j];
    }
    z[t] = 1.0f / (acc + 1e-5f);
}

// ---------------------------------------------------------------------------
// ctx[b,s,h*64+m] = (q2[b,s,h,:] @ kv[b,h,:,:]) * z[b,s,h]
// grid (B*H, S/64). kv (16 KB) staged transposed into LDS once per block;
// A fragments read straight from global (q2 rows are K-contiguous).
// ---------------------------------------------------------------------------
__global__ __launch_bounds__(256) void attn_kernel(const bf16_t* __restrict__ q2,
                                                   const bf16_t* __restrict__ kv,
                                                   const float* __restrict__ z,
                                                   bf16_t* __restrict__ ctx) {
    __shared__ bf16_t kvT[64 * 128];  // [m][kk]  16 KB
    const int bh = blockIdx.x;
    const int b = bh / Hc, h = bh % Hc;
    const int s0 = blockIdx.y * 64;
    const int tid = threadIdx.x, lane = tid & 31, wave = tid >> 5;
    const int wS = (wave & 3) * 16;   // s sub-tile
    const int wN = (wave >> 2) * 32;  // m cols 0 / 32

    {   // stage kv transposed: [128 kk][64 m] -> kvT[m][kk]
        const int kk = tid >> 1;
        const int mc = (tid & 1) * 32;
        const bf16_t* kp = kv + (size_t)bh * 8192 + kk * 64 + mc;
#pragma unroll
        for (int c = 0; c < 4; ++c) {
            union { uint4 u; __bf16 hh[8]; } uu;
            uu.u = *(const uint4*)(kp + c * 8);
#pragma unroll
            for (int j = 0; j < 8; ++j)
                kvT[(mc + c * 8 + j) * 128 + kk] = uu.hh[j];
        }
    }
    __syncthreads();

    v8f acc[2] = {};
#pragma unroll
    for (int k0 = 0; k0 < 128; k0 += 32) {
        const int srow = s0 + wS + (lane & 15);
        const bf16_t* ap =
            q2 + ((size_t)(b * Sc + srow) * Hc + h) * 128 + k0 + (lane >> 4) * 8;
        v16bf a = frag_load_contig(ap);

        v16bf bfrag[2];
#pragma unroll
        for (int jt = 0; jt < 2; ++jt)
            bfrag[jt] = frag_load_contig(frag_ptr(kvT, 128, wN + jt * 16, k0, lane));
#pragma unroll
        for (int jt = 0; jt < 2; ++jt)
            acc[jt] = WMMA_BF16(a, bfrag[jt], acc[jt]);
    }

    const int nOut  = lane & 15;
    const int mBase = (lane >> 4) * 8;
#pragma unroll
    for (int jt = 0; jt < 2; ++jt)
#pragma unroll
        for (int r = 0; r < 8; ++r) {
            const int s = s0 + wS + mBase + r;
            const float zz = z[(size_t)(b * Sc + s) * Hc + h];
            ctx[(size_t)(b * Sc + s) * Dc + h * DHc + wN + jt * 16 + nOut] =
                (bf16_t)(acc[jt][r] * zz);
        }
}

// ---------------------------------------------------------------------------
// Host-side launch
// ---------------------------------------------------------------------------
extern "C" void kernel_launch(void* const* d_in, const int* in_sizes, int n_in,
                              void* d_out, int out_size, void* d_ws, size_t ws_size,
                              hipStream_t stream) {
    (void)in_sizes; (void)n_in; (void)out_size; (void)ws_size;

    const float*         x    = (const float*)d_in[0];
    const unsigned char* mask = (const unsigned char*)d_in[1];
    const float* cosb = (const float*)d_in[2];
    const float* sinb = (const float*)d_in[3];
    const float* g1   = (const float*)d_in[4];
    const float* be1  = (const float*)d_in[5];
    const float* g2   = (const float*)d_in[6];
    const float* be2  = (const float*)d_in[7];
    const float* Wq = (const float*)d_in[8],  *bq = (const float*)d_in[9];
    const float* Wk = (const float*)d_in[10], *bk = (const float*)d_in[11];
    const float* Wv = (const float*)d_in[12], *bv = (const float*)d_in[13];
    const float* Wo = (const float*)d_in[14], *bo = (const float*)d_in[15];
    const float* W1 = (const float*)d_in[16], *b1 = (const float*)d_in[17];
    const float* W2 = (const float*)d_in[18], *b2 = (const float*)d_in[19];

    // Bump allocator over workspace (256B aligned chunks).
    uintptr_t wsp = (uintptr_t)d_ws;
    auto alloc = [&](size_t bytes) -> void* {
        uintptr_t p = (wsp + 255) & ~(uintptr_t)255;
        wsp = p + bytes;
        return (void*)p;
    };

    bf16_t* wqT = (bf16_t*)alloc((size_t)Dc * Dc * 2);
    bf16_t* wkT = (bf16_t*)alloc((size_t)Dc * Dc * 2);
    bf16_t* wvT = (bf16_t*)alloc((size_t)Dc * Dc * 2);
    bf16_t* woT = (bf16_t*)alloc((size_t)Dc * Dc * 2);
    bf16_t* w1T = (bf16_t*)alloc((size_t)Dc * FFc * 2);   // [FF][D]
    bf16_t* w2T = (bf16_t*)alloc((size_t)FFc * Dc * 2);   // [D][FF]
    bf16_t* xn  = (bf16_t*)alloc((size_t)BS * Dc * 2);    // also hn
    bf16_t* qraw = (bf16_t*)alloc((size_t)BS * Dc * 2);   // h1 reuses from here
    bf16_t* kraw = (bf16_t*)alloc((size_t)BS * Dc * 2);
    bf16_t* q2  = (bf16_t*)alloc((size_t)BS * Hc * 128 * 2);
    bf16_t* k2  = (bf16_t*)alloc((size_t)BS * Hc * 128 * 2);
    bf16_t* vbf = (bf16_t*)alloc((size_t)BS * Dc * 2);
    bf16_t* ctx = (bf16_t*)alloc((size_t)BS * Dc * 2);
    bf16_t* kvb = (bf16_t*)alloc((size_t)Bc * Hc * 128 * 64 * 2);
    float*  ksum = (float*)alloc((size_t)Bc * Hc * 128 * 4);
    float*  zb   = (float*)alloc((size_t)BS * Hc * 4);
    float*  ybuf = (float*)alloc((size_t)BS * Dc * 4);
    bf16_t* h1 = qraw;  // qraw+kraw+q2 region (exactly BS*3072 bf16) reused for FFN hidden

    // 1) weight conversion + transpose (W[K][N] f32 -> WT[N][K] bf16)
    auto convT = [&](const float* src, bf16_t* dst, int K, int N) {
        conv_transpose_kernel<<<dim3(N / 32, K / 32), 256, 0, stream>>>(src, dst, K, N);
    };
    convT(Wq, wqT, Dc, Dc); convT(Wk, wkT, Dc, Dc); convT(Wv, wvT, Dc, Dc);
    convT(Wo, woT, Dc, Dc); convT(W1, w1T, Dc, FFc); convT(W2, w2T, FFc, Dc);

    // 2) LN1
    ln_kernel<<<BS, 256, 0, stream>>>(x, g1, be1, xn, Dc);

    // 3) Q/K/V projections (bias fused)
    dim3 gD(Dc / 128, BS / 128);
    gemm_bias_act_kernel<<<gD, 256, 0, stream>>>(xn, wqT, bq, nullptr, nullptr, qraw, BS, Dc, Dc, 0);
    gemm_bias_act_kernel<<<gD, 256, 0, stream>>>(xn, wkT, bk, nullptr, nullptr, kraw, BS, Dc, Dc, 0);
    gemm_bias_act_kernel<<<gD, 256, 0, stream>>>(xn, wvT, bv, nullptr, nullptr, vbf,  BS, Dc, Dc, 0);

    // 4) relu/mask + cos-sin reweight
    {
        int total = BS * Hc * DHc;
        reweight_kernel<<<(total + 255) / 256, 256, 0, stream>>>(qraw, kraw, cosb, sinb,
                                                                 mask, q2, k2, total);
    }

    // 5) kv (+ksum), z
    kv_kernel<<<Bc * Hc, 256, 0, stream>>>(k2, vbf, kvb, ksum);
    {
        int total = BS * Hc;
        z_kernel<<<(total + 255) / 256, 256, 0, stream>>>(q2, ksum, zb, total);
    }

    // 6) attention output per head
    attn_kernel<<<dim3(Bc * Hc, Sc / 64), 256, 0, stream>>>(q2, kvb, zb, ctx);

    // 7) Wo projection + residual -> y (f32)
    gemm_bias_act_kernel<<<gD, 256, 0, stream>>>(ctx, woT, bo, x, ybuf, nullptr, BS, Dc, Dc, 0);

    // 8) LN2 -> hn (reuse xn buffer)
    ln_kernel<<<BS, 256, 0, stream>>>(ybuf, g2, be2, xn, Dc);

    // 9) FFN up + GELU
    gemm_bias_act_kernel<<<dim3(FFc / 128, BS / 128), 256, 0, stream>>>(
        xn, w1T, b1, nullptr, nullptr, h1, BS, FFc, Dc, 1);

    // 10) FFN down + residual -> out (f32)
    gemm_bias_act_kernel<<<gD, 256, 0, stream>>>(
        h1, w2T, b2, ybuf, (float*)d_out, nullptr, BS, Dc, FFc, 0);
}
